// AttLayer_42193758716084
// MI455X (gfx1250) — matrix operand order repaired and checked by reading, not compiled, not run on hardware
//
#include <hip/hip_runtime.h>

typedef __attribute__((ext_vector_type(16))) _Float16 v16h;
typedef __attribute__((ext_vector_type(8)))  float    v8f;

#define BSZ    64
#define TSZ    4096
#define DSZ    256
#define ASZ    50          // attention dim (padded to 64 columns)
#define TTILE  128         // rows of x per block
#define NTILES (TSZ / TTILE)   // 32
#define EPSV   1e-7f

// ---------------------------------------------------------------------------
// Pass 1: one block = (batch b, 128-row tile). 256 threads = 8 waves.
//   Wave w computes scores for rows [t0 + 16w, t0 + 16w + 16) via WMMA,
//   then the whole block does the fp32 weighted column sums (L2-hot re-read).
// ---------------------------------------------------------------------------
__global__ __launch_bounds__(256) void att_pass1(
    const float* __restrict__ x,     // [B, T, D]
    const float* __restrict__ W,     // [D, A]
    const float* __restrict__ bias,  // [A]
    const float* __restrict__ u,     // [A]
    float* __restrict__ Ppart,       // [B, NTILES, D] unnormalized partial sums
    float* __restrict__ Spart)       // [B, NTILES]    partial score sums
{
    // B fragments of W in f16, pre-packed in wave32 WMMA layout:
    // lane = {hi half selects K range, lo 4 bits select column}, 16 halves = K.
    __shared__ __align__(32) _Float16 Bf[4][8][32][16];   // 32 KB
    __shared__ float u_s[64];
    __shared__ float b_s[64];
    __shared__ float s_all[TTILE];

    const int tid  = threadIdx.x;
    const int bb   = blockIdx.y;
    const int tile = blockIdx.x;
    const int t0   = tile * TTILE;

    // ---- pre-pack W (f32 -> f16) into B-fragment layout, pad cols 50..63 ---
    for (int idx = tid; idx < 4 * 8 * 32 * 16; idx += 256) {
        const int h    = idx & 15;
        const int lane = (idx >> 4) & 31;
        const int kb   = (idx >> 9) & 7;
        const int nt   = idx >> 12;
        const int k    = kb * 32 + (h < 8 ? (lane >> 4) * 8 + h : 16 + (lane >> 4) * 8 + (h - 8));   // K index (0..255)
        const int n    = nt * 16 + (lane & 15);            // column (0..63)
        Bf[nt][kb][lane][h] = (n < ASZ) ? (_Float16)W[k * ASZ + n]
                                        : (_Float16)0.0f;
    }
    if (tid < 64) {
        u_s[tid] = (tid < ASZ) ? u[tid]    : 0.0f;
        b_s[tid] = (tid < ASZ) ? bias[tid] : 0.0f;
    }
    __syncthreads();

    const int w    = tid >> 5;       // wave id (0..7), 16 rows each
    const int lane = tid & 31;
    const int lo   = lane & 15;
    const int hi   = lane >> 4;

    // ---- load A fragments: 16 rows x 256 K of x, f32 -> f16 ----------------
    // 16-bit A layout: lane lo = row M; hi-half lanes hold K%16 in [8,16).
    const float* xrow = x + ((size_t)bb * TSZ + (t0 + w * 16 + lo)) * DSZ;
    v16h a[8];
#pragma unroll
    for (int kb = 0; kb < 8; ++kb) {
        const int k0 = kb * 32 + hi * 8;
        const float4 f0 = *(const float4*)(xrow + k0);
        const float4 f1 = *(const float4*)(xrow + k0 + 4);
        const float4 f2 = *(const float4*)(xrow + k0 + 16);
        const float4 f3 = *(const float4*)(xrow + k0 + 20);
        v16h av;
        av[0]  = (_Float16)f0.x; av[1]  = (_Float16)f0.y;
        av[2]  = (_Float16)f0.z; av[3]  = (_Float16)f0.w;
        av[4]  = (_Float16)f1.x; av[5]  = (_Float16)f1.y;
        av[6]  = (_Float16)f1.z; av[7]  = (_Float16)f1.w;
        av[8]  = (_Float16)f2.x; av[9]  = (_Float16)f2.y;
        av[10] = (_Float16)f2.z; av[11] = (_Float16)f2.w;
        av[12] = (_Float16)f3.x; av[13] = (_Float16)f3.y;
        av[14] = (_Float16)f3.z; av[15] = (_Float16)f3.w;
        a[kb] = av;
    }

    // ---- GEMM tile: uit_pre = x_tile (16x256) * W (256x64) -----------------
    v8f c[4] = {};
#pragma unroll
    for (int nt = 0; nt < 4; ++nt) {
#pragma unroll
        for (int kb = 0; kb < 8; ++kb) {
            const v16h bv = *(const v16h*)(&Bf[nt][kb][lane][0]);
            c[nt] = __builtin_amdgcn_wmma_f32_16x16x32_f16(
                false, a[kb], false, bv, (short)0, c[nt], false, false);
        }
    }

    // ---- scores: s = exp( sum_n tanh(c + bias[n]) * u[n] ) -----------------
    // C layout: VGPR v holds row M=v (lanes 0-15) / M=v+8 (lanes 16-31),
    // N = nt*16 + lo. Butterfly over 16 lanes stays within each half.
#pragma unroll
    for (int v = 0; v < 8; ++v) {
        float p = 0.0f;
#pragma unroll
        for (int nt = 0; nt < 4; ++nt) {
            const int col = nt * 16 + lo;
            p += tanhf(c[nt][v] + b_s[col]) * u_s[col];
        }
        p += __shfl_xor(p, 1, 32);
        p += __shfl_xor(p, 2, 32);
        p += __shfl_xor(p, 4, 32);
        p += __shfl_xor(p, 8, 32);
        const float s = __expf(p);
        if (lo == v) s_all[w * 16 + hi * 8 + v] = s;   // two writers: hi=0,1
    }
    __syncthreads();

    // ---- per-tile score sum (deterministic, single writer) -----------------
    if (tid == 0) {
        float t = 0.0f;
        for (int r = 0; r < TTILE; ++r) t += s_all[r];
        Spart[bb * NTILES + tile] = t;
    }

    // ---- weighted column sums: acc[d] = sum_r s[r] * x[r][d] (fp32, L2-hot)
    const float* xblk = x + ((size_t)bb * TSZ + t0) * DSZ;
    float acc = 0.0f;
#pragma unroll 4
    for (int r = 0; r < TTILE; ++r)
        acc += s_all[r] * xblk[(size_t)r * DSZ + tid];
    Ppart[((size_t)bb * NTILES + tile) * DSZ + tid] = acc;
}

// ---------------------------------------------------------------------------
// Pass 2: reduce tile partials and normalize. One block per batch.
// ---------------------------------------------------------------------------
__global__ __launch_bounds__(256) void att_pass2(
    const float* __restrict__ Ppart,
    const float* __restrict__ Spart,
    float* __restrict__ out)
{
    const int bb = blockIdx.x;
    const int d  = threadIdx.x;

    float ss = 0.0f;
#pragma unroll 8
    for (int t = 0; t < NTILES; ++t) ss += Spart[bb * NTILES + t];

    float acc = 0.0f;
#pragma unroll 8
    for (int t = 0; t < NTILES; ++t)
        acc += Ppart[((size_t)bb * NTILES + t) * DSZ + d];

    out[bb * DSZ + d] = acc / (ss + EPSV);
}

extern "C" void kernel_launch(void* const* d_in, const int* in_sizes, int n_in,
                              void* d_out, int out_size, void* d_ws, size_t ws_size,
                              hipStream_t stream)
{
    const float* x    = (const float*)d_in[0];   // [64, 4096, 256]
    const float* W    = (const float*)d_in[1];   // [256, 50]
    const float* bias = (const float*)d_in[2];   // [50]
    const float* u    = (const float*)d_in[3];   // [50, 1]
    float* out        = (float*)d_out;           // [64, 256]

    float* Ppart = (float*)d_ws;                       // 64*32*256 floats
    float* Spart = Ppart + (size_t)BSZ * NTILES * DSZ; // 64*32 floats

    att_pass1<<<dim3(NTILES, BSZ), 256, 0, stream>>>(x, W, bias, u, Ppart, Spart);
    att_pass2<<<BSZ, DSZ, 0, stream>>>(Ppart, Spart, out);
}
